// MultiHeadAttention_35897336660750
// MI455X (gfx1250) — matrix operand order repaired and checked by reading, not compiled, not run on hardware
//
#include <hip/hip_runtime.h>

// ---------------------------------------------------------------------------
// Causal multi-head attention forward for MI455X (gfx1250, wave32, WMMA).
// B=4, T=2048, C=1024, H=16, D=64.  All matmuls on v_wmma_f32_16x16x32_bf16.
// Double-buffered LDS GEMM with async global->LDS copies when available.
// ---------------------------------------------------------------------------

#define NH   16
#define HD   64
#define CE   1024            // n_embed
#define BB   4
#define TT   2048
#define MTOT (BB * TT)       // 8192 rows
#define NQKV (3 * CE)        // 3072
#define LDST 40              // padded LDS row stride (bf16 elems): 80B = 5 banks

#ifndef __has_builtin
#define __has_builtin(x) 0
#endif
#if __has_builtin(__builtin_amdgcn_global_load_async_to_lds_b128) && \
    __has_builtin(__builtin_amdgcn_s_wait_asynccnt)
#define HAVE_ASYNC_LDS 1
#else
#define HAVE_ASYNC_LDS 0
#endif

typedef __attribute__((ext_vector_type(16))) __bf16 bf16x16;
typedef __attribute__((ext_vector_type(8)))  float  f32x8;

// Exact match for the async-to-LDS builtin's parameter types:
// '__attribute__((__vector_size__(4 * sizeof(int)))) int __device__ *'
typedef int v4i_vs __attribute__((vector_size(16)));
typedef __attribute__((address_space(1))) v4i_vs* global_v4i_ptr;
typedef __attribute__((address_space(3))) v4i_vs* lds_v4i_ptr;

union FragBF {
    bf16x16 v;
    uint4   u[2];
    unsigned short s[16];
};
union FragF {
    f32x8 v;
    float f[8];
};

__device__ __forceinline__ unsigned short f32_to_bf16(float f) {
    union { __bf16 b; unsigned short s; } u;
    u.b = (__bf16)f;            // hardware f32->bf16 convert (RTNE)
    return u.s;
}

__device__ __forceinline__ f32x8 wmma_bf16(const bf16x16& a, const bf16x16& b, const f32x8& c) {
    // D = A(16x32 bf16) * B(32x16 bf16) + C(16x16 f32)
    return __builtin_amdgcn_wmma_f32_16x16x32_bf16(
        /*neg_a=*/false, a, /*neg_b=*/false, b,
        /*c_mod=*/(short)0, c, /*reuse_a=*/false, /*reuse_b=*/false);
}

// ---------------------------------------------------------------------------
// Kernel 0: fp32 -> bf16 flat convert (for x).
// ---------------------------------------------------------------------------
__global__ void __launch_bounds__(256)
convert_f32_to_bf16_kernel(const float* __restrict__ in,
                           unsigned short* __restrict__ out) {
    size_t i = ((size_t)blockIdx.x * 256 + threadIdx.x) * 4;
    float4 f = *(const float4*)(in + i);
    ushort4 o;
    o.x = f32_to_bf16(f.x);
    o.y = f32_to_bf16(f.y);
    o.z = f32_to_bf16(f.z);
    o.w = f32_to_bf16(f.w);
    *(ushort4*)(out + i) = o;
}

// ---------------------------------------------------------------------------
// Kernel 1: fp32 [K][N] -> bf16 [N][K] transpose (for weight matrices).
// ---------------------------------------------------------------------------
__global__ void __launch_bounds__(256)
transpose_f32_to_bf16_kernel(const float* __restrict__ in,
                             unsigned short* __restrict__ out,
                             int K, int N) {
    __shared__ float tile[32][33];
    const int n0 = blockIdx.x * 32;
    const int k0 = blockIdx.y * 32;
    const int tx = threadIdx.x & 31;
    const int ty = threadIdx.x >> 5;            // 0..7
    #pragma unroll
    for (int i = ty; i < 32; i += 8)
        tile[i][tx] = in[(size_t)(k0 + i) * N + n0 + tx];
    __syncthreads();
    #pragma unroll
    for (int i = ty; i < 32; i += 8)
        out[(size_t)(n0 + i) * K + k0 + tx] = f32_to_bf16(tile[tx][i]);
}

// ---------------------------------------------------------------------------
// Tile staging: 128 rows x 32 bf16 -> LDS (padded stride LDST).
// 512 b128 chunks / 256 threads = 2 per thread. Async when available.
// ---------------------------------------------------------------------------
__device__ __forceinline__ void stage_tile(const unsigned short* __restrict__ g,
                                           int ld, int row0, int k0,
                                           unsigned short* lds) {
    const int tid = threadIdx.x;
    #pragma unroll
    for (int i = 0; i < 2; ++i) {
        int c   = tid + i * 256;
        int row = c >> 2;
        int col = (c & 3) * 8;
        const unsigned short* src = g + (size_t)(row0 + row) * ld + k0 + col;
        unsigned short*       dst = lds + row * LDST + col;
#if HAVE_ASYNC_LDS
        __builtin_amdgcn_global_load_async_to_lds_b128(
            (global_v4i_ptr)(src),
            (lds_v4i_ptr)(dst),
            /*offset=*/0, /*cpol=*/0);
#else
        *(uint4*)dst = *(const uint4*)src;
#endif
    }
}

// ---------------------------------------------------------------------------
// Shared GEMM mainloop: C[128x128] = A_bf16[M][K] * Bt_bf16[N][K]^T.
// Block = 256 threads = 8 waves, wave grid 4(M) x 2(N), wave tile 32x64.
// Double-buffered LDS, one barrier per 32-K step, 8 WMMAs/wave/step.
// ---------------------------------------------------------------------------
__device__ __forceinline__ void gemm_mainloop(const unsigned short* __restrict__ A,
                                              const unsigned short* __restrict__ Bt,
                                              int lda, int ldb, int K,
                                              int m0, int n0,
                                              FragF acc[2][4]) {
    __shared__ unsigned short As[2][128 * LDST];
    __shared__ unsigned short Bs[2][128 * LDST];

    const int lane  = threadIdx.x & 31;
    const int wid   = threadIdx.x >> 5;
    const int waveM = wid & 3;                  // 0..3
    const int waveN = wid >> 2;                 // 0..1
    const int lhalf = lane >> 4;                // 0/1
    const int l16   = lane & 15;

    #pragma unroll
    for (int mt = 0; mt < 2; ++mt)
        #pragma unroll
        for (int nt = 0; nt < 4; ++nt)
            #pragma unroll
            for (int e = 0; e < 8; ++e) acc[mt][nt].f[e] = 0.0f;

    // Prologue: stage first tile.
    stage_tile(A,  lda, m0, 0, As[0]);
    stage_tile(Bt, ldb, n0, 0, Bs[0]);
#if HAVE_ASYNC_LDS
    __builtin_amdgcn_s_wait_asynccnt(0);
#endif
    __syncthreads();

    const int nsteps = K / 32;
    for (int s = 0; s < nsteps; ++s) {
        const int cur = s & 1;
        const int nxt = cur ^ 1;
        if (s + 1 < nsteps) {
            stage_tile(A,  lda, m0, (s + 1) * 32, As[nxt]);
            stage_tile(Bt, ldb, n0, (s + 1) * 32, Bs[nxt]);
        }

        FragBF a[2], b[4];
        #pragma unroll
        for (int mt = 0; mt < 2; ++mt) {
            // A-frag: lane row = l16; K elems: lhalf*8+[0..8) and 16+lhalf*8+[0..8)
            const unsigned short* base = As[cur] + (waveM * 32 + mt * 16 + l16) * LDST;
            a[mt].u[0] = *(const uint4*)(base + lhalf * 8);
            a[mt].u[1] = *(const uint4*)(base + 16 + lhalf * 8);
        }
        #pragma unroll
        for (int nt = 0; nt < 4; ++nt) {
            // B-frag: lane col = l16; K elems: lhalf*8+[0..8) and 16+lhalf*8+[0..8)
            const unsigned short* base =
                Bs[cur] + (waveN * 64 + nt * 16 + l16) * LDST;
            b[nt].u[0] = *(const uint4*)(base + lhalf * 8);
            b[nt].u[1] = *(const uint4*)(base + 16 + lhalf * 8);
        }
        #pragma unroll
        for (int mt = 0; mt < 2; ++mt)
            #pragma unroll
            for (int nt = 0; nt < 4; ++nt)
                acc[mt][nt].v = wmma_bf16(a[mt].v, b[nt].v, acc[mt][nt].v);

#if HAVE_ASYNC_LDS
        if (s + 1 < nsteps) __builtin_amdgcn_s_wait_asynccnt(0);
#endif
        __syncthreads();
    }
}

// ---------------------------------------------------------------------------
// Kernel 2: QKV projection. qkv[m][n] = xb[m][:] . w_qkv[:][n],
// scattered to q[B,H,T,D], k[B,H,T,D] (bf16) and vT[B,H,D,T] (bf16).
// Section (q/k/v) is block-uniform: scalar branch, no exec divergence.
// ---------------------------------------------------------------------------
__global__ void __launch_bounds__(256)
gemm_qkv_kernel(const unsigned short* __restrict__ xb,
                const unsigned short* __restrict__ wqkvT,
                unsigned short* __restrict__ q,
                unsigned short* __restrict__ k,
                unsigned short* __restrict__ vT) {
    const int m0 = blockIdx.y * 128;
    const int n0 = blockIdx.x * 128;
    FragF acc[2][4];
    gemm_mainloop(xb, wqkvT, CE, CE, CE, m0, n0, acc);

    const int lane  = threadIdx.x & 31;
    const int wid   = threadIdx.x >> 5;
    const int waveM = wid & 3, waveN = wid >> 2;
    const int lhalf = lane >> 4, l16 = lane & 15;
    const int sec   = n0 >> 10;                 // block-uniform: 0=q, 1=k, 2=v

    if (sec < 2) {
        unsigned short* dstbuf = (sec == 0) ? q : k;
        #pragma unroll
        for (int mt = 0; mt < 2; ++mt)
            #pragma unroll
            for (int nt = 0; nt < 4; ++nt)
                #pragma unroll
                for (int e = 0; e < 8; ++e) {
                    int m = m0 + waveM * 32 + mt * 16 + lhalf * 8 + e;
                    int n = n0 + waveN * 64 + nt * 16 + l16;
                    int b = m >> 11, t = m & (TT - 1);
                    int c = n & (CE - 1);
                    int h = c >> 6, d = c & (HD - 1);
                    size_t bh = (size_t)(b * NH + h);
                    dstbuf[(bh * TT + t) * HD + d] = f32_to_bf16(acc[mt][nt].f[e]);
                }
    } else {
        #pragma unroll
        for (int mt = 0; mt < 2; ++mt)
            #pragma unroll
            for (int nt = 0; nt < 4; ++nt)
                #pragma unroll
                for (int e = 0; e < 8; ++e) {
                    int m = m0 + waveM * 32 + mt * 16 + lhalf * 8 + e;
                    int n = n0 + waveN * 64 + nt * 16 + l16;
                    int b = m >> 11, t = m & (TT - 1);
                    int c = n & (CE - 1);
                    int h = c >> 6, d = c & (HD - 1);
                    size_t bh = (size_t)(b * NH + h);
                    vT[(bh * HD + d) * TT + t] = f32_to_bf16(acc[mt][nt].f[e]);
                }
    }
}

// ---------------------------------------------------------------------------
// Kernel 3: causal flash attention. One 16-query tile per wave; 32-key steps.
// S = (Q Kᵀ)·scale, online softmax, O += P V.  ctx written bf16 [B,T,C].
// ---------------------------------------------------------------------------
__global__ void __launch_bounds__(256)
flash_attn_kernel(const unsigned short* __restrict__ q,
                  const unsigned short* __restrict__ kk,
                  const unsigned short* __restrict__ vT,
                  unsigned short* __restrict__ ctx) {
    const int bh = blockIdx.y;                  // 0..B*NH-1
    const int b  = bh / NH, h = bh % NH;
    const int lane  = threadIdx.x & 31;
    const int wid   = threadIdx.x >> 5;
    const int lhalf = lane >> 4, l16 = lane & 15;
    const int q0 = blockIdx.x * 128 + wid * 16; // query tile start

    const unsigned short* qb = q  + (size_t)bh * TT * HD;
    const unsigned short* kb = kk + (size_t)bh * TT * HD;
    const unsigned short* vb = vT + (size_t)bh * HD * TT;

    __shared__ unsigned short Ps[8][16 * LDST]; // per-wave P scratch (bf16)
    unsigned short* ps = &Ps[wid][0];

    // Q fragments (d = 0..31, 32..63), loaded once.
    FragBF qa[2];
    #pragma unroll
    for (int dk = 0; dk < 2; ++dk) {
        const unsigned short* base = qb + (size_t)(q0 + l16) * HD + dk * 32;
        qa[dk].u[0] = *(const uint4*)(base + lhalf * 8);
        qa[dk].u[1] = *(const uint4*)(base + 16 + lhalf * 8);
    }

    FragF o[4];                                 // O: d tiles of 16
    float mrow[8], lrow[8];
    #pragma unroll
    for (int nt = 0; nt < 4; ++nt)
        #pragma unroll
        for (int e = 0; e < 8; ++e) o[nt].f[e] = 0.0f;
    #pragma unroll
    for (int e = 0; e < 8; ++e) { mrow[e] = -1e30f; lrow[e] = 0.0f; }

    const float scale = 0.125f;                 // 1/sqrt(64)
    const int   kend  = q0 + 16;                // causal: keys in [0, kend)

    for (int kt = 0; kt < kend; kt += 32) {
        // Prefetch next key/value slab into cache (global_prefetch_b8).
        if (kt + 32 < kend) {
            __builtin_prefetch(kb + (size_t)(kt + 32 + l16) * HD, 0, 3);
            __builtin_prefetch(vb + (size_t)l16 * TT + kt + 32, 0, 3);
        }

        // ---- S = Q Kᵀ over a 16x32 key slab (two 16x16 C frags) ----
        FragF s[2];
        #pragma unroll
        for (int st = 0; st < 2; ++st)
            #pragma unroll
            for (int e = 0; e < 8; ++e) s[st].f[e] = 0.0f;

        #pragma unroll
        for (int st = 0; st < 2; ++st)
            #pragma unroll
            for (int dk = 0; dk < 2; ++dk) {
                FragBF bk;   // B-frag: col = key (l16), K = d (lhalf*8+e, 16+lhalf*8+e)
                const unsigned short* base =
                    kb + (size_t)(kt + st * 16 + l16) * HD + dk * 32;
                bk.u[0] = *(const uint4*)(base + lhalf * 8);
                bk.u[1] = *(const uint4*)(base + 16 + lhalf * 8);
                s[st].v = wmma_bf16(qa[dk].v, bk.v, s[st].v);
            }

        // ---- scale + causal mask ----
        #pragma unroll
        for (int st = 0; st < 2; ++st)
            #pragma unroll
            for (int e = 0; e < 8; ++e) {
                int qi = q0 + lhalf * 8 + e;
                int ki = kt + st * 16 + l16;
                float xv = s[st].f[e] * scale;
                s[st].f[e] = (ki <= qi) ? xv : -1e30f;
            }

        // ---- online softmax: row max across 32 cols (16-lane halves) ----
        float mnew[8];
        #pragma unroll
        for (int e = 0; e < 8; ++e) {
            float xv = fmaxf(s[0].f[e], s[1].f[e]);
            #pragma unroll
            for (int off = 1; off < 16; off <<= 1)
                xv = fmaxf(xv, __shfl_xor(xv, off, 32));
            mnew[e] = fmaxf(mrow[e], xv);
        }
        #pragma unroll
        for (int e = 0; e < 8; ++e) {
            float alpha = __expf(mrow[e] - mnew[e]);
            lrow[e] *= alpha;
            #pragma unroll
            for (int nt = 0; nt < 4; ++nt) o[nt].f[e] *= alpha;
            mrow[e] = mnew[e];
        }
        #pragma unroll
        for (int st = 0; st < 2; ++st)
            #pragma unroll
            for (int e = 0; e < 8; ++e)
                s[st].f[e] = __expf(s[st].f[e] - mrow[e]);
        #pragma unroll
        for (int e = 0; e < 8; ++e) {
            float xv = s[0].f[e] + s[1].f[e];
            #pragma unroll
            for (int off = 1; off < 16; off <<= 1)
                xv += __shfl_xor(xv, off, 32);
            lrow[e] += xv;
        }

        // ---- re-layout P (C-frag -> A-frag) through per-wave LDS ----
        #pragma unroll
        for (int st = 0; st < 2; ++st)
            #pragma unroll
            for (int e = 0; e < 8; ++e)
                ps[(lhalf * 8 + e) * LDST + st * 16 + l16] = f32_to_bf16(s[st].f[e]);
        __threadfence_block();                  // order wave-private LDS RAW

        FragBF pa;
        {
            const unsigned short* base = ps + l16 * LDST;
            pa.u[0] = *(const uint4*)(base + lhalf * 8);
            pa.u[1] = *(const uint4*)(base + 16 + lhalf * 8);
        }
        __threadfence_block();                  // protect scratch before next iter

        // ---- O += P V  (V read transposed: contiguous along key) ----
        #pragma unroll
        for (int nt = 0; nt < 4; ++nt) {
            FragBF bv;   // col = d (nt*16+l16), K = key (kt + lhalf*8 + e, kt + 16 + lhalf*8 + e)
            const unsigned short* base =
                vb + (size_t)(nt * 16 + l16) * TT + kt;
            bv.u[0] = *(const uint4*)(base + lhalf * 8);
            bv.u[1] = *(const uint4*)(base + 16 + lhalf * 8);
            o[nt].v = wmma_bf16(pa.v, bv.v, o[nt].v);
        }
    }

    // ---- normalize and write ctx [B][T][C] bf16 ----
    #pragma unroll
    for (int nt = 0; nt < 4; ++nt)
        #pragma unroll
        for (int e = 0; e < 8; ++e) {
            int qi = q0 + lhalf * 8 + e;
            int d  = nt * 16 + l16;
            ctx[((size_t)b * TT + qi) * CE + h * HD + d] =
                f32_to_bf16(o[nt].f[e] / lrow[e]);
        }
}

// ---------------------------------------------------------------------------
// Kernel 4: output projection out = ctx @ w_out + b_out  (fp32 result).
// ---------------------------------------------------------------------------
__global__ void __launch_bounds__(256)
gemm_out_kernel(const unsigned short* __restrict__ ctx,
                const unsigned short* __restrict__ woutT,
                const float* __restrict__ bias,
                float* __restrict__ out) {
    const int m0 = blockIdx.y * 128;
    const int n0 = blockIdx.x * 128;
    FragF acc[2][4];
    gemm_mainloop(ctx, woutT, CE, CE, CE, m0, n0, acc);

    const int lane  = threadIdx.x & 31;
    const int wid   = threadIdx.x >> 5;
    const int waveM = wid & 3, waveN = wid >> 2;
    const int lhalf = lane >> 4, l16 = lane & 15;

    float bvals[4];
    #pragma unroll
    for (int nt = 0; nt < 4; ++nt)
        bvals[nt] = bias[n0 + waveN * 64 + nt * 16 + l16];

    #pragma unroll
    for (int mt = 0; mt < 2; ++mt)
        #pragma unroll
        for (int nt = 0; nt < 4; ++nt)
            #pragma unroll
            for (int e = 0; e < 8; ++e) {
                int m = m0 + waveM * 32 + mt * 16 + lhalf * 8 + e;
                int n = n0 + waveN * 64 + nt * 16 + l16;
                out[(size_t)m * CE + n] = acc[mt][nt].f[e] + bvals[nt];
            }
}

// ---------------------------------------------------------------------------
// Host-side orchestration.
// ---------------------------------------------------------------------------
extern "C" void kernel_launch(void* const* d_in, const int* in_sizes, int n_in,
                              void* d_out, int out_size, void* d_ws, size_t ws_size,
                              hipStream_t stream) {
    const float* x     = (const float*)d_in[0];   // [B,T,C]
    const float* w_qkv = (const float*)d_in[1];   // [C, 3C]
    const float* w_out = (const float*)d_in[2];   // [C, C]
    const float* b_out = (const float*)d_in[3];   // [C]
    float* out = (float*)d_out;                   // [B,T,C] fp32

    // Workspace layout (bytes):
    char* ws = (char*)d_ws;
    size_t off = 0;
    unsigned short* xb    = (unsigned short*)(ws + off); off += (size_t)MTOT * CE * 2;  // 16 MB
    unsigned short* wqkvT = (unsigned short*)(ws + off); off += (size_t)NQKV * CE * 2;  //  6 MB
    unsigned short* woutT = (unsigned short*)(ws + off); off += (size_t)CE * CE * 2;    //  2 MB
    unsigned short* qbuf  = (unsigned short*)(ws + off); off += (size_t)MTOT * CE * 2;  // 16 MB
    unsigned short* kbuf  = (unsigned short*)(ws + off); off += (size_t)MTOT * CE * 2;  // 16 MB
    unsigned short* vTbuf = (unsigned short*)(ws + off); off += (size_t)MTOT * CE * 2;  // 16 MB
    unsigned short* ctx   = (unsigned short*)(ws + off); off += (size_t)MTOT * CE * 2;  // 16 MB

    // 0) x -> bf16
    convert_f32_to_bf16_kernel<<<dim3((MTOT * CE) / 1024), 256, 0, stream>>>(x, xb);

    // 1) weight transposes to bf16 [N][K]
    transpose_f32_to_bf16_kernel<<<dim3(NQKV / 32, CE / 32), 256, 0, stream>>>(
        w_qkv, wqkvT, CE, NQKV);
    transpose_f32_to_bf16_kernel<<<dim3(CE / 32, CE / 32), 256, 0, stream>>>(
        w_out, woutT, CE, CE);

    // 2) QKV projection
    gemm_qkv_kernel<<<dim3(NQKV / 128, MTOT / 128), 256, 0, stream>>>(
        xb, wqkvT, qbuf, kbuf, vTbuf);

    // 3) causal flash attention
    flash_attn_kernel<<<dim3(TT / 128, BB * NH), 256, 0, stream>>>(
        qbuf, kbuf, vTbuf, ctx);

    // 4) output projection + bias
    gemm_out_kernel<<<dim3(CE / 128, MTOT / 128), 256, 0, stream>>>(
        ctx, woutT, b_out, out);
}
